// VeRAAdapter_52836687675852
// MI455X (gfx1250) — matrix operand order repaired and checked by reading, not compiled, not run on hardware
//
#include <hip/hip_runtime.h>

// ---------------------------------------------------------------------------
// VeRA adapter, MI455X (gfx1250, wave32, WMMA bf16 path)
//   out = x @ W^T + bias + ((x @ A^T) * d) @ (diag(b) B)^T
// f32->bf16 once into d_ws (operands L2-resident: 96 MB < 192 MB L2), then
// two bf16 WMMA GEMMs. Double-buffered LDS tiles filled with
// GLOBAL_LOAD_ASYNC_TO_LDS_B128 (ASYNCcnt), one barrier per K-step.
// ---------------------------------------------------------------------------

#define NROWS 8192
#define INF   4096
#define OUTF  4096
#define RANKF 256
#define LDSTR 40   // padded LDS row stride in bf16 elems (80B, 16B aligned, conflict-free)

typedef __bf16 bf16_t;
typedef bf16_t v16bf __attribute__((ext_vector_type(16)));
typedef float  v8f   __attribute__((ext_vector_type(8)));
typedef int    v4i   __attribute__((ext_vector_type(4)));
typedef unsigned int   u32x4 __attribute__((ext_vector_type(4)));
typedef unsigned short u16x4 __attribute__((ext_vector_type(4)));

union FragAB { v16bf v; u32x4 u[2]; };

#if __has_builtin(__builtin_amdgcn_global_load_async_to_lds_b128)
#define HAS_ASYNC_LDS 1
typedef __attribute__((address_space(1))) v4i* g_v4i_p;   // global int4*
typedef __attribute__((address_space(3))) v4i* l_v4i_p;   // LDS int4*
#endif

// copy 16 bytes (8 bf16) global -> LDS
__device__ __forceinline__ void copy16_g2l(const unsigned short* g, unsigned short* l) {
#ifdef HAS_ASYNC_LDS
  __builtin_amdgcn_global_load_async_to_lds_b128((g_v4i_p)(v4i*)g, (l_v4i_p)(v4i*)l, 0, 0);
#else
  *(u32x4*)l = *(const u32x4*)g;
#endif
}

__device__ __forceinline__ void wait_tiles() {
#ifdef HAS_ASYNC_LDS
#if __has_builtin(__builtin_amdgcn_s_wait_asynccnt)
  __builtin_amdgcn_s_wait_asynccnt(0);
#else
  asm volatile("s_wait_asynccnt 0x0" ::: "memory");
#endif
#endif
}

__device__ __forceinline__ unsigned short f2bf_rne(float f) {
  unsigned int u = __float_as_uint(f);
  u += 0x7FFFu + ((u >> 16) & 1u);          // round-to-nearest-even
  return (unsigned short)(u >> 16);
}

// ---------------- f32 -> bf16 bulk conversion (vec4) -----------------------
__global__ __launch_bounds__(256) void cvt_f32_bf16_vec4(
    const float* __restrict__ in, unsigned short* __restrict__ out, long long n4) {
  long long i = (long long)blockIdx.x * blockDim.x + threadIdx.x;
  long long stride = (long long)gridDim.x * blockDim.x;
  for (; i < n4; i += stride) {
    float4 f = ((const float4*)in)[i];
    u16x4 o = { f2bf_rne(f.x), f2bf_rne(f.y), f2bf_rne(f.z), f2bf_rne(f.w) };
    ((u16x4*)out)[i] = o;
  }
}

// ---------------- B' = diag(b_vec) * B, converted to bf16 ------------------
__global__ __launch_bounds__(256) void scale_convert_B(
    const float* __restrict__ B, const float* __restrict__ bvec,
    unsigned short* __restrict__ out, int n4) {  // n4 = OUTF*RANKF/4
  int i = blockIdx.x * blockDim.x + threadIdx.x;
  int stride = gridDim.x * blockDim.x;
  for (; i < n4; i += stride) {
    float4 f = ((const float4*)B)[i];
    float s = bvec[i >> 6];                  // (i*4)/RANKF, RANKF=256
    u16x4 o = { f2bf_rne(f.x * s), f2bf_rne(f.y * s),
                f2bf_rne(f.z * s), f2bf_rne(f.w * s) };
    ((u16x4*)out)[i] = o;
  }
}

// ---------------- H = (x @ A^T) * d  -> bf16 [NROWS][RANKF] ----------------
// block tile 128M x 64N, 8 waves (4Mx2N), wave tile 32x32 (2x2 frags)
// double-buffered LDS, async tile fill
__global__ __launch_bounds__(256) void vera_h_kernel(
    const unsigned short* __restrict__ xb, const unsigned short* __restrict__ ab,
    const float* __restrict__ dvec, unsigned short* __restrict__ hb) {
  __shared__ unsigned short sA[2][128 * LDSTR];
  __shared__ unsigned short sB[2][64 * LDSTR];

  const int tid = threadIdx.x, lane = tid & 31, wv = tid >> 5;
  const int wm = wv >> 1, wn = wv & 1, l16 = lane & 15, lh = lane >> 4;

  const unsigned short* Ag = xb + (size_t)(blockIdx.y * 128) * INF;
  const unsigned short* Bg = ab + (size_t)(blockIdx.x * 64) * INF;

  // per-thread staging assignments
  const int arow = tid >> 1, aseg = tid & 1;       // A: 16 elems/thread
  const int brow = tid >> 2, bseg = tid & 3;       // B: 8 elems/thread

  auto issue_tile = [&](int kb, int buf) {
    const unsigned short* ga = Ag + (size_t)arow * INF + kb * 32 + aseg * 16;
    unsigned short* la = sA[buf] + arow * LDSTR + aseg * 16;
    copy16_g2l(ga, la);
    copy16_g2l(ga + 8, la + 8);
    const unsigned short* gb = Bg + (size_t)brow * INF + kb * 32 + bseg * 8;
    unsigned short* lb = sB[buf] + brow * LDSTR + bseg * 8;
    copy16_g2l(gb, lb);
  };

  const v8f vzero = {0.f, 0.f, 0.f, 0.f, 0.f, 0.f, 0.f, 0.f};
  v8f acc[2][2];
  for (int i = 0; i < 2; ++i)
    for (int j = 0; j < 2; ++j) acc[i][j] = vzero;

  const int KS = INF / 32;
  issue_tile(0, 0);
  int buf = 0;
  for (int kb = 0; kb < KS; ++kb) {
    wait_tiles();
    __syncthreads();
    if (kb + 1 < KS) issue_tile(kb + 1, buf ^ 1);

    FragAB af[2], bfr[2];
    for (int mi = 0; mi < 2; ++mi) {
      const unsigned short* p = sA[buf] + (wm * 32 + mi * 16 + l16) * LDSTR;
      af[mi].u[0] = *(const u32x4*)(p + 8 * lh);        // K 0..7 / 8..15
      af[mi].u[1] = *(const u32x4*)(p + 16 + 8 * lh);   // K 16..23 / 24..31
    }
    for (int ni = 0; ni < 2; ++ni) {
      const unsigned short* p = sB[buf] + (wn * 32 + ni * 16 + l16) * LDSTR;
      bfr[ni].u[0] = *(const u32x4*)(p + 8 * lh);
      bfr[ni].u[1] = *(const u32x4*)(p + 16 + 8 * lh);
    }
    for (int mi = 0; mi < 2; ++mi)
      for (int ni = 0; ni < 2; ++ni)
        acc[mi][ni] = __builtin_amdgcn_wmma_f32_16x16x32_bf16(
            false, af[mi].v, false, bfr[ni].v, (short)0, acc[mi][ni], false, false);
    buf ^= 1;
  }

  const int gm0 = blockIdx.y * 128 + wm * 32;
  const int gn0 = blockIdx.x * 64 + wn * 32;
  for (int ni = 0; ni < 2; ++ni) {
    int gn = gn0 + ni * 16 + l16;
    float dv = dvec[gn];
    for (int mi = 0; mi < 2; ++mi) {
      int gmb = gm0 + mi * 16 + 8 * lh;
      for (int v = 0; v < 8; ++v)
        hb[(size_t)(gmb + v) * RANKF + gn] = f2bf_rne(acc[mi][ni][v] * dv);
    }
  }
}

// ------- out = x@W^T + H@B'^T + bias  (fused as K = 4096 then K = 256) -----
// block tile 128M x 256N, 8 waves (2Mx4N), wave tile 64x64 (4x4 frags)
// both K-phases flattened into one 136-step double-buffered pipeline
__global__ __launch_bounds__(256) void vera_main_kernel(
    const unsigned short* __restrict__ xb, const unsigned short* __restrict__ wb,
    const unsigned short* __restrict__ hb, const unsigned short* __restrict__ bpb,
    const float* __restrict__ bias, float* __restrict__ out) {
  __shared__ unsigned short sA[2][128 * LDSTR];
  __shared__ unsigned short sB[2][256 * LDSTR];

  const int tid = threadIdx.x, lane = tid & 31, wv = tid >> 5;
  const int wm = wv >> 2, wn = wv & 3, l16 = lane & 15, lh = lane >> 4;

  const unsigned short* Aph[2] = { xb + (size_t)(blockIdx.y * 128) * INF,
                                   hb + (size_t)(blockIdx.y * 128) * RANKF };
  const unsigned short* Bph[2] = { wb  + (size_t)(blockIdx.x * 256) * INF,
                                   bpb + (size_t)(blockIdx.x * 256) * RANKF };

  const int arow = tid >> 1, aseg = tid & 1;       // A: 16 elems/thread

  auto issue_tile = [&](int it, int buf) {
    const int ph = (it < INF / 32) ? 1 : 0;        // 1 => phase0 (x/W)
    const unsigned short* Ag = ph ? Aph[0] : Aph[1];
    const unsigned short* Bg = ph ? Bph[0] : Bph[1];
    const int ld = ph ? INF : RANKF;
    const int kb = ph ? it : it - INF / 32;
    const unsigned short* ga = Ag + (size_t)arow * ld + kb * 32 + aseg * 16;
    unsigned short* la = sA[buf] + arow * LDSTR + aseg * 16;
    copy16_g2l(ga, la);
    copy16_g2l(ga + 8, la + 8);
    const unsigned short* gb = Bg + (size_t)tid * ld + kb * 32;  // B: one row/thread
    unsigned short* lb = sB[buf] + tid * LDSTR;
    copy16_g2l(gb, lb);
    copy16_g2l(gb + 8, lb + 8);
    copy16_g2l(gb + 16, lb + 16);
    copy16_g2l(gb + 24, lb + 24);
  };

  const v8f vzero = {0.f, 0.f, 0.f, 0.f, 0.f, 0.f, 0.f, 0.f};
  v8f acc[4][4];
  for (int i = 0; i < 4; ++i)
    for (int j = 0; j < 4; ++j) acc[i][j] = vzero;

  const int TOT = INF / 32 + RANKF / 32;   // 136 K-steps
  issue_tile(0, 0);
  int buf = 0;
  for (int it = 0; it < TOT; ++it) {
    wait_tiles();
    __syncthreads();
    if (it + 1 < TOT) issue_tile(it + 1, buf ^ 1);

    FragAB af[4], bfr[4];
    for (int mi = 0; mi < 4; ++mi) {
      const unsigned short* p = sA[buf] + (wm * 64 + mi * 16 + l16) * LDSTR;
      af[mi].u[0] = *(const u32x4*)(p + 8 * lh);
      af[mi].u[1] = *(const u32x4*)(p + 16 + 8 * lh);
    }
    for (int ni = 0; ni < 4; ++ni) {
      const unsigned short* p = sB[buf] + (wn * 64 + ni * 16 + l16) * LDSTR;
      bfr[ni].u[0] = *(const u32x4*)(p + 8 * lh);
      bfr[ni].u[1] = *(const u32x4*)(p + 16 + 8 * lh);
    }
    for (int mi = 0; mi < 4; ++mi)
      for (int ni = 0; ni < 4; ++ni)
        acc[mi][ni] = __builtin_amdgcn_wmma_f32_16x16x32_bf16(
            false, af[mi].v, false, bfr[ni].v, (short)0, acc[mi][ni], false, false);
    buf ^= 1;
  }

  // epilogue: + bias, f32 store
  const int gm0 = blockIdx.y * 128 + wm * 64;
  const int gn0 = blockIdx.x * 256 + wn * 64;
  for (int ni = 0; ni < 4; ++ni) {
    int gn = gn0 + ni * 16 + l16;
    float bb = bias[gn];
    for (int mi = 0; mi < 4; ++mi) {
      int gmb = gm0 + mi * 16 + 8 * lh;
      for (int v = 0; v < 8; ++v)
        out[(size_t)(gmb + v) * OUTF + gn] = acc[mi][ni][v] + bb;
    }
  }
}

// ---------------------------------------------------------------------------
extern "C" void kernel_launch(void* const* d_in, const int* in_sizes, int n_in,
                              void* d_out, int out_size, void* d_ws, size_t ws_size,
                              hipStream_t stream) {
  const float* x    = (const float*)d_in[0];
  const float* W    = (const float*)d_in[1];
  const float* bias = (const float*)d_in[2];
  const float* A    = (const float*)d_in[3];
  const float* B    = (const float*)d_in[4];
  const float* bvec = (const float*)d_in[5];
  const float* dvec = (const float*)d_in[6];
  float* out = (float*)d_out;

  // workspace layout (bf16 as ushort): x | W | A | B' | H  (~104 MB total)
  unsigned short* xb  = (unsigned short*)d_ws;
  unsigned short* wb  = xb  + (size_t)NROWS * INF;
  unsigned short* ab  = wb  + (size_t)OUTF  * INF;
  unsigned short* bpb = ab  + (size_t)RANKF * INF;
  unsigned short* hb  = bpb + (size_t)OUTF  * RANKF;

  cvt_f32_bf16_vec4<<<4096, 256, 0, stream>>>(x, xb, (long long)NROWS * INF / 4);
  cvt_f32_bf16_vec4<<<4096, 256, 0, stream>>>(W, wb, (long long)OUTF * INF / 4);
  cvt_f32_bf16_vec4<<<1024, 256, 0, stream>>>(A, ab, (long long)RANKF * INF / 4);
  scale_convert_B<<<1024, 256, 0, stream>>>(B, bvec, bpb, OUTF * RANKF / 4);

  vera_h_kernel<<<dim3(RANKF / 64, NROWS / 128), 256, 0, stream>>>(xb, ab, dvec, hb);
  vera_main_kernel<<<dim3(OUTF / 256, NROWS / 128), 256, 0, stream>>>(
      xb, wb, hb, bpb, bias, out);
}
